// MultiHeadAttention_9354438771333
// MI455X (gfx1250) — matrix operand order repaired and checked by reading, not compiled, not run on hardware
//
#include <hip/hip_runtime.h>

// ---------------------------------------------------------------------------
// Multi-head attention for MI455X (gfx1250), wave32 + WMMA f32_16x16x32_f16.
// Phases: cast->f16, QKV GEMM (WMMA), flash attention (WMMA + online softmax
// + GLOBAL_LOAD_ASYNC_TO_LDS staging), projection GEMM (WMMA).
// ---------------------------------------------------------------------------

typedef unsigned int  u32;
typedef unsigned short u16;
typedef unsigned long long u64;
typedef __attribute__((ext_vector_type(16))) _Float16 v16h;
typedef __attribute__((ext_vector_type(8)))  float    v8f;
typedef __attribute__((ext_vector_type(8)))  u32      v8u;

#define D_MODEL   1024
#define NUM_HEADS 16
#define HEAD_DIM  64
#define BATCH     2
#define SEQ       2048
#define M_TOTAL   (BATCH * SEQ)      // 4096 token rows
#define QK_SCALE  0.125f             // 64^-0.5
#define LOG2E     1.4426950408889634f

union FragU { v8u u; v16h h; };

__device__ inline v8f wmma16(v16h a, v16h b, v8f c) {
  // D = A(16x32 f16) * B(32x16 f16) + C(16x16 f32)
  return __builtin_amdgcn_wmma_f32_16x16x32_f16(
      /*neg_a=*/false, a, /*neg_b=*/false, b,
      /*c_mod=*/(short)0, c, /*reuse_a=*/false, /*reuse_b=*/false);
}

__device__ inline u16 f2h_bits(float f) {
  _Float16 h = (_Float16)f;
  return __builtin_bit_cast(u16, h);
}

// Async memory->LDS copy, 16 bytes per lane (tracked by ASYNCcnt).
// lds_byte_off: LDS byte address (low 32 bits of generic shared address,
// per ISA 10.2 LDS aperture truncation).  gaddr: 64-bit global address.
__device__ inline void async_load_lds_b128(u32 lds_byte_off, u64 gaddr) {
#if defined(__AMDGCN__)
  asm volatile("global_load_async_to_lds_b128 %0, %1, off"
               :: "v"(lds_byte_off), "v"(gaddr) : "memory");
#else
  (void)lds_byte_off; (void)gaddr;
#endif
}

__device__ inline void wait_asynccnt0() {
#if defined(__AMDGCN__)
  asm volatile("s_wait_asynccnt 0x0" ::: "memory");
#endif
}

// A fragment: 16x32 f16 tile from row-major src (leading dim ld, elements).
// ISA layout: lane row = row0 + lane%16; half = lane/16;
//   vgpr i<4  -> K = k0 + half*8 + 2*i   .. +1
//   vgpr i>=4 -> K = k0 + 16 + half*8 + 2*(i-4) .. +1
__device__ inline v16h load_frag_a(const u16* base, int ld, int row0, int k0, int lane) {
  FragU f;
  int row  = row0 + (lane & 15);
  int half = lane >> 4;
  const u32* p = (const u32*)(base + (size_t)row * ld);
  #pragma unroll
  for (int i = 0; i < 8; ++i) {
    int k = k0 + ((i < 4) ? 0 : 16) + half * 8 + 2 * (i & 3);
    f.u[i] = p[k >> 1];
  }
  return f.h;
}

// B fragment: 32x16 where B[k][n] = W[n][k]; W row-major, leading dim ld.
// ISA layout: lane col n = n0 + lane%16; half = lane/16;
//   vgpr i<4  -> K = k0 + half*8 + 2*i .. +1
//   vgpr i>=4 -> K = k0 + 16 + half*8 + 2*(i-4) .. +1
__device__ inline v16h load_frag_b(const u16* base, int ld, int n0, int k0, int lane) {
  FragU f;
  int n  = n0 + (lane & 15);
  int kb = (k0 + (lane >> 4) * 8) >> 1;    // in u32 units
  const u32* p = (const u32*)(base + (size_t)n * ld);
  #pragma unroll
  for (int i = 0; i < 8; ++i) f.u[i] = p[kb + ((i < 4) ? i : i + 4)];
  return f.h;
}

// ---------------------------------------------------------------------------
// Phase 1: f32 -> f16 cast
// ---------------------------------------------------------------------------
__global__ void cast_f32_to_f16(const float* __restrict__ src, u16* __restrict__ dst, int n) {
  int i = blockIdx.x * blockDim.x + threadIdx.x;
  if (i < n) dst[i] = f2h_bits(src[i]);
}

// ---------------------------------------------------------------------------
// Phase 2: QKV GEMM.  qkv[m][c] = sum_k x[m][k]*w_qkv[c][k] + b_qkv[c]
// Scatter into head-major Q/K/V [b][h][n][d]; Q pre-scaled by scale*log2e.
// One wave -> 16x64 output tile.
// ---------------------------------------------------------------------------
__global__ __launch_bounds__(256)
void gemm_qkv_kernel(const u16* __restrict__ X, const u16* __restrict__ W,
                     const float* __restrict__ bias,
                     u16* __restrict__ Qo, u16* __restrict__ Ko, u16* __restrict__ Vo) {
  int gw   = (blockIdx.x * blockDim.x + threadIdx.x) >> 5;
  int lane = threadIdx.x & 31;
  const int NT = (3 * D_MODEL) / 64;          // 48 column tiles
  int mt = gw / NT, nt = gw % NT;
  int m0 = mt * 16, n0 = nt * 64;

  v8f c[4] = {};
  for (int k0 = 0; k0 < D_MODEL; k0 += 32) {
    // prefetch next K-step of the streamed operands into cache
    __builtin_prefetch(&X[(size_t)(m0 + (lane & 15)) * D_MODEL + k0 + 64], 0, 1);
    __builtin_prefetch(&W[(size_t)(n0 + (lane & 15)) * D_MODEL + k0 + 64], 0, 1);
    v16h a = load_frag_a(X, D_MODEL, m0, k0, lane);
    #pragma unroll
    for (int t = 0; t < 4; ++t)
      c[t] = wmma16(a, load_frag_b(W, D_MODEL, n0 + t * 16, k0, lane), c[t]);
  }

  int half = lane >> 4;
  #pragma unroll
  for (int t = 0; t < 4; ++t) {
    int col = n0 + t * 16 + (lane & 15);
    int sel = col >> 10;                 // 0=Q 1=K 2=V
    int rem = col & (D_MODEL - 1);
    int h   = rem >> 6;
    int d   = rem & (HEAD_DIM - 1);
    u16* dst = (sel == 0) ? Qo : ((sel == 1) ? Ko : Vo);
    float bv = bias[col];
    float sc = (sel == 0) ? (QK_SCALE * LOG2E) : 1.0f;
    #pragma unroll
    for (int r = 0; r < 8; ++r) {
      int m = m0 + r + 8 * half;         // global token row
      int b = m >> 11;                   // / SEQ
      int n = m & (SEQ - 1);
      float v = (c[t][r] + bv) * sc;
      dst[(((size_t)(b * NUM_HEADS + h)) * SEQ + n) * HEAD_DIM + d] = f2h_bits(v);
    }
  }
}

// ---------------------------------------------------------------------------
// Phase 3: flash attention.  Block = 8 waves sharing one (b,h); each wave owns
// 16 query rows; K tile (32x64) async-DMA'd into LDS, V^T tile (64x32) staged
// with a register transpose.  Online softmax in exp2 domain (scale*log2e
// already folded into Q).
// ---------------------------------------------------------------------------
__global__ __launch_bounds__(256)
void attn_kernel(const u16* __restrict__ Q, const u16* __restrict__ K,
                 const u16* __restrict__ V, u16* __restrict__ O) {
  __shared__ __align__(16) u16 sK[32 * HEAD_DIM];       // [key][d] 4 KB
  __shared__ __align__(16) u16 sVt[HEAD_DIM][34];       // [d][key] padded
  __shared__ __align__(16) u16 sP[8][16 * 32];          // per-wave P tile

  const int MG = SEQ / 128;                 // 16 query groups per (b,h)
  int bh   = blockIdx.x / MG;
  int mg   = blockIdx.x % MG;
  int wave = threadIdx.x >> 5;
  int lane = threadIdx.x & 31;
  int half = lane >> 4;
  int m0   = mg * 128 + wave * 16;

  const u16* Qb = Q + (size_t)bh * SEQ * HEAD_DIM;
  const u16* Kb = K + (size_t)bh * SEQ * HEAD_DIM;
  const u16* Vb = V + (size_t)bh * SEQ * HEAD_DIM;

  v16h qa0 = load_frag_a(Qb, HEAD_DIM, m0, 0, lane);
  v16h qa1 = load_frag_a(Qb, HEAD_DIM, m0, 32, lane);

  v8f o0 = {}, o1 = {}, o2 = {}, o3 = {};
  float mrow[8], lrow[8];
  #pragma unroll
  for (int r = 0; r < 8; ++r) { mrow[r] = -1e30f; lrow[r] = 0.0f; }

  u32 sK_lds = (u32)(uintptr_t)&sK[0] + threadIdx.x * 16u;

  for (int j0 = 0; j0 < SEQ; j0 += 32) {
    // K tile (32x64 f16 = 4 KB): async DMA, 256 lanes x 16 B
    async_load_lds_b128(sK_lds,
        (u64)(uintptr_t)(Kb + (size_t)j0 * HEAD_DIM) + (u64)threadIdx.x * 16u);
    // V tile transposed into sVt via registers (DMA cannot reshuffle)
    {
      union { uint4 v; u16 s[8]; } tv;
      tv.v = ((const uint4*)(Vb + (size_t)j0 * HEAD_DIM))[threadIdx.x];
      int key = threadIdx.x >> 3;
      int d0  = (threadIdx.x & 7) * 8;
      #pragma unroll
      for (int e = 0; e < 8; ++e) sVt[d0 + e][key] = tv.s[e];
    }
    wait_asynccnt0();
    __syncthreads();

    // S = Q (16x64) * K^T -> two 16x16 tiles (keys j0..j0+15, j0+16..j0+31)
    v8f s0 = {}, s1 = {};
    s0 = wmma16(qa0, load_frag_b(sK, HEAD_DIM, 0,  0,  lane), s0);
    s0 = wmma16(qa1, load_frag_b(sK, HEAD_DIM, 0,  32, lane), s0);
    s1 = wmma16(qa0, load_frag_b(sK, HEAD_DIM, 16, 0,  lane), s1);
    s1 = wmma16(qa1, load_frag_b(sK, HEAD_DIM, 16, 32, lane), s1);

    u16* pw = sP[wave];
    #pragma unroll
    for (int r = 0; r < 8; ++r) {
      float tm = fmaxf(s0[r], s1[r]);
      #pragma unroll
      for (int d = 1; d < 16; d <<= 1) tm = fmaxf(tm, __shfl_xor(tm, d, 32));
      float mn    = fmaxf(mrow[r], tm);
      float alpha = exp2f(mrow[r] - mn);
      mrow[r] = mn;
      float p0 = exp2f(s0[r] - mn);
      float p1 = exp2f(s1[r] - mn);
      float rs = p0 + p1;
      #pragma unroll
      for (int d = 1; d < 16; d <<= 1) rs += __shfl_xor(rs, d, 32);
      lrow[r] = lrow[r] * alpha + rs;
      int row = r + 8 * half, col = lane & 15;
      pw[row * 32 + col]      = f2h_bits(p0);
      pw[row * 32 + col + 16] = f2h_bits(p1);
      o0[r] *= alpha; o1[r] *= alpha; o2[r] *= alpha; o3[r] *= alpha;
    }

    // O += P (16x32) * V (32x64)
    v16h pa = load_frag_a(pw, 32, 0, 0, lane);
    o0 = wmma16(pa, load_frag_b(&sVt[0][0], 34, 0,  0, lane), o0);
    o1 = wmma16(pa, load_frag_b(&sVt[0][0], 34, 16, 0, lane), o1);
    o2 = wmma16(pa, load_frag_b(&sVt[0][0], 34, 32, 0, lane), o2);
    o3 = wmma16(pa, load_frag_b(&sVt[0][0], 34, 48, 0, lane), o3);
    __syncthreads();
  }

  // epilogue: normalize and store f16 into [b][n][h*64+d] layout
  int b = bh / NUM_HEADS, h = bh % NUM_HEADS;
  int cl = lane & 15;
  #pragma unroll
  for (int r = 0; r < 8; ++r) {
    float inv = 1.0f / lrow[r];
    int n = m0 + r + 8 * half;
    size_t base = ((size_t)(b * SEQ + n)) * D_MODEL + h * HEAD_DIM;
    O[base +  0 + cl] = f2h_bits(o0[r] * inv);
    O[base + 16 + cl] = f2h_bits(o1[r] * inv);
    O[base + 32 + cl] = f2h_bits(o2[r] * inv);
    O[base + 48 + cl] = f2h_bits(o3[r] * inv);
  }
}

// ---------------------------------------------------------------------------
// Phase 4: projection GEMM. out[m][c] = sum_k a[m][k]*w_proj[c][k] + b_proj[c]
// ---------------------------------------------------------------------------
__global__ __launch_bounds__(256)
void gemm_proj_kernel(const u16* __restrict__ A, const u16* __restrict__ W,
                      const float* __restrict__ bias, float* __restrict__ out) {
  int gw   = (blockIdx.x * blockDim.x + threadIdx.x) >> 5;
  int lane = threadIdx.x & 31;
  const int NT = D_MODEL / 64;                // 16 column tiles
  int mt = gw / NT, nt = gw % NT;
  int m0 = mt * 16, n0 = nt * 64;

  v8f c[4] = {};
  for (int k0 = 0; k0 < D_MODEL; k0 += 32) {
    __builtin_prefetch(&A[(size_t)(m0 + (lane & 15)) * D_MODEL + k0 + 64], 0, 1);
    __builtin_prefetch(&W[(size_t)(n0 + (lane & 15)) * D_MODEL + k0 + 64], 0, 1);
    v16h a = load_frag_a(A, D_MODEL, m0, k0, lane);
    #pragma unroll
    for (int t = 0; t < 4; ++t)
      c[t] = wmma16(a, load_frag_b(W, D_MODEL, n0 + t * 16, k0, lane), c[t]);
  }

  int half = lane >> 4;
  #pragma unroll
  for (int t = 0; t < 4; ++t) {
    int col = n0 + t * 16 + (lane & 15);
    float bv = bias[col];
    #pragma unroll
    for (int r = 0; r < 8; ++r) {
      int m = m0 + r + 8 * half;
      out[(size_t)m * D_MODEL + col] = c[t][r] + bv;
    }
  }
}

// ---------------------------------------------------------------------------
// Launcher
// ---------------------------------------------------------------------------
extern "C" void kernel_launch(void* const* d_in, const int* in_sizes, int n_in,
                              void* d_out, int out_size, void* d_ws, size_t ws_size,
                              hipStream_t stream) {
  (void)in_sizes; (void)n_in; (void)out_size; (void)ws_size;

  const float* x      = (const float*)d_in[0];
  const float* w_qkv  = (const float*)d_in[1];
  const float* b_qkv  = (const float*)d_in[2];
  const float* w_proj = (const float*)d_in[3];
  const float* b_proj = (const float*)d_in[4];
  float* out = (float*)d_out;

  // workspace layout (f16 halves), total ~50 MB
  u16* ws = (u16*)d_ws;
  size_t off = 0;
  u16* xh     = ws + off; off += (size_t)M_TOTAL * D_MODEL;        // 4096x1024
  u16* wqkvh  = ws + off; off += (size_t)3 * D_MODEL * D_MODEL;    // 3072x1024
  u16* wprojh = ws + off; off += (size_t)D_MODEL * D_MODEL;        // 1024x1024
  u16* Qh     = ws + off; off += (size_t)M_TOTAL * D_MODEL;        // [b][h][n][d]
  u16* Kh     = ws + off; off += (size_t)M_TOTAL * D_MODEL;
  u16* Vh     = ws + off; off += (size_t)M_TOTAL * D_MODEL;
  u16* AOh    = ws + off; off += (size_t)M_TOTAL * D_MODEL;        // attn out

  const int T = 256;
  int n_x  = M_TOTAL * D_MODEL;
  int n_wq = 3 * D_MODEL * D_MODEL;
  int n_wp = D_MODEL * D_MODEL;
  cast_f32_to_f16<<<(n_x  + T - 1) / T, T, 0, stream>>>(x,      xh,     n_x);
  cast_f32_to_f16<<<(n_wq + T - 1) / T, T, 0, stream>>>(w_qkv,  wqkvh,  n_wq);
  cast_f32_to_f16<<<(n_wp + T - 1) / T, T, 0, stream>>>(w_proj, wprojh, n_wp);

  // QKV: (4096/16) m-tiles * (3072/64) n-tiles = 12288 waves / 8 per block
  gemm_qkv_kernel<<<(256 * 48) / 8, T, 0, stream>>>(xh, wqkvh, b_qkv, Qh, Kh, Vh);

  // attention: B*H*(SEQ/128) = 512 blocks of 8 waves
  attn_kernel<<<BATCH * NUM_HEADS * (SEQ / 128), T, 0, stream>>>(Qh, Kh, Vh, AOh);

  // projection: (4096/16)*(1024/64) = 4096 waves / 8 per block
  gemm_proj_kernel<<<(256 * 16) / 8, T, 0, stream>>>(AOh, wprojh, b_proj, out);
}
